// DilatedMHCABlock_53412213293416
// MI455X (gfx1250) — compile-verified, matrix operand order checked by reading, not run on hardware
//
#include <hip/hip_runtime.h>

// ---------------------------------------------------------------------------
// Types for CDNA5 WMMA (wave32, 16x16x32 bf16 -> f32)
// ---------------------------------------------------------------------------
typedef __attribute__((ext_vector_type(16))) __bf16 v16bf;
typedef __attribute__((ext_vector_type(8)))  __bf16 v8bf;
typedef __attribute__((ext_vector_type(4)))  __bf16 v4bf;
typedef __attribute__((ext_vector_type(8)))  float  v8f;

#define D_MODEL 1024
#define N_HEADS 16
#define D_HEAD  64
#define SEQ_N   2048
#define BATCH   4
#define ROWS    (BATCH * SEQ_N)          // 8192
#define NTOK    ((size_t)ROWS * D_MODEL) // 8,388,608

// ---------------------------------------------------------------------------
// gfx1250 async copy: 16 bytes global -> LDS, tracked by ASYNCcnt.
// VDST VGPR carries the LDS byte address (low 32 bits of generic pointer).
// ---------------------------------------------------------------------------
__device__ __forceinline__ void async_cp16(unsigned lds_off,
                                           const __bf16* __restrict__ g) {
    asm volatile("global_load_async_to_lds_b128 %0, %1, off"
                 :: "v"(lds_off), "v"(g)
                 : "memory");
}

__device__ __forceinline__ void wait_async0() {
    asm volatile("s_wait_asynccnt 0x0" ::: "memory");
}

// ---------------------------------------------------------------------------
// Fragment load: per-lane 16 bf16 of one A row / one B column.
// ISA 16-bit A-matrix 16x32 layout: lane L holds m=L%16; with half=L/16,
// elems 0..7 from K = half*8 + e, elems 8..15 from K = 16 + half*8 + (e-8).
// Two contiguous 16-byte chunks (works for global or LDS pointers).
// ---------------------------------------------------------------------------
__device__ __forceinline__ v16bf load_frag(const __bf16* p, int half) {
    union { v16bf v; v8bf h[2]; } u;
    u.h[0] = *(const v8bf*)(p + half * 8);
    u.h[1] = *(const v8bf*)(p + 16 + half * 8);
    return u.v;
}

// ---------------------------------------------------------------------------
// f32 -> bf16 conversion (4 elements / thread)
// ---------------------------------------------------------------------------
__global__ void to_bf16_kernel(const float4* __restrict__ src,
                               __bf16* __restrict__ dst, int n4) {
    int i = blockIdx.x * blockDim.x + threadIdx.x;
    if (i < n4) {
        float4 v = src[i];
        v4bf o;
        o.x = (__bf16)v.x; o.y = (__bf16)v.y;
        o.z = (__bf16)v.z; o.w = (__bf16)v.w;
        *(v4bf*)(dst + (size_t)i * 4) = o;
    }
}

// ---------------------------------------------------------------------------
// One k-step of the fused QKV tile: preload all 12 B fragments into distinct
// register sets, then stream the 12 WMMAs.  sched_group_barrier requests the
// pattern [2 VMEM-read][24 DS-read][12 WMMA] so DS reads batch under a single
// dscnt wait instead of one LDS round-trip per WMMA.
// ---------------------------------------------------------------------------
__device__ __forceinline__ void qkv_step(const __bf16* __restrict__ arow_k,
                                         const __bf16 (*sBbuf)[64][32],
                                         int mr, int half, v8f acc[3][4]) {
    v16bf af = load_frag(arow_k, half);
    v16bf bfr[4][3];
    #pragma unroll
    for (int t = 0; t < 4; ++t)
        #pragma unroll
        for (int o = 0; o < 3; ++o)
            bfr[t][o] = load_frag(&sBbuf[o][t * 16 + mr][0], half);
    #pragma unroll
    for (int t = 0; t < 4; ++t)
        #pragma unroll
        for (int o = 0; o < 3; ++o)
            acc[o][t] = __builtin_amdgcn_wmma_f32_16x16x32_bf16(
                false, af, false, bfr[t][o], (short)0, acc[o][t], false, false);

    // Scheduling hint: A global reads, then all LDS fragment reads, then WMMAs
    __builtin_amdgcn_sched_group_barrier(0x0020, 2, 0);   // VMEM read x2 (A)
    __builtin_amdgcn_sched_group_barrier(0x0100, 24, 0);  // DS read x24 (B)
    __builtin_amdgcn_sched_group_barrier(0x0008, 12, 0);  // WMMA x12
}

// ---------------------------------------------------------------------------
// Fused QKV projection: q/k/v = x @ W{q,k,v}.T + b
// Block (256 thr, 8 waves) computes a 128-row x 64-col tile of q,k,v.
// B panels (3 x 64col x 32K bf16 = 12KB) staged in double-buffered LDS via
// async copies, shared by all 8 waves; A fragments direct from global.
// k-loop manually unrolled x2 so buffer parity is a compile-time constant.
// __launch_bounds__(256,1): allow full VGPR budget for 12 live fragments.
// grid = (16 col strips, 64 row blocks), block = 256.
// ---------------------------------------------------------------------------
__global__ __launch_bounds__(256, 1) void qkv_gemm_kernel(
    const __bf16* __restrict__ xb,
    const __bf16* __restrict__ wq, const __bf16* __restrict__ wk,
    const __bf16* __restrict__ wv,
    const float* __restrict__ bq, const float* __restrict__ bk,
    const float* __restrict__ bv,
    float* __restrict__ qf, float* __restrict__ kf, float* __restrict__ vf) {

    __shared__ __bf16 sB[2][3][64][32];          // 24 KB

    const int tid  = threadIdx.x;
    const int lane = tid & 31;
    const int wave = tid >> 5;
    const int colbase = blockIdx.x * 64;
    const int row0 = blockIdx.y * 128 + wave * 16;
    const int mr   = lane & 15;
    const int half = lane >> 4;

    const __bf16* __restrict__ w[3] = { wq, wk, wv };
    const float*  __restrict__ bias[3] = { bq, bk, bv };
    float* __restrict__ outp[3] = { qf, kf, vf };

    // Staging: thread stages one 16B chunk per weight per k-step:
    // col = tid>>2 (0..63), cc = tid&3 (16B chunk of the 64B K-row).
    const int scol = tid >> 2;
    const int scc  = tid & 3;
    const __bf16* gsrc[3];
    unsigned ldst0[3], ldst1[3];
    #pragma unroll
    for (int o = 0; o < 3; ++o) {
        gsrc[o]  = w[o] + (size_t)(colbase + scol) * D_MODEL + scc * 8;
        ldst0[o] = (unsigned)(uintptr_t)&sB[0][o][scol][scc * 8];
        ldst1[o] = (unsigned)(uintptr_t)&sB[1][o][scol][scc * 8];
    }

    const __bf16* __restrict__ arow = xb + (size_t)(row0 + mr) * D_MODEL;

    v8f acc[3][4] = {};

    // Prologue: stage k-step 0 into buffer 0
    #pragma unroll
    for (int o = 0; o < 3; ++o) async_cp16(ldst0[o], gsrc[o]);

    for (int i = 0; i < 32; i += 2) {
        // ---- step i (buffer 0) ----
        wait_async0();
        __syncthreads();
        #pragma unroll
        for (int o = 0; o < 3; ++o)                       // i+1 <= 31 always
            async_cp16(ldst1[o], gsrc[o] + (i + 1) * 32);
        qkv_step(arow + i * 32, sB[0], mr, half, acc);
        __syncthreads();

        // ---- step i+1 (buffer 1) ----
        wait_async0();
        __syncthreads();
        if (i + 2 < 32) {
            #pragma unroll
            for (int o = 0; o < 3; ++o)
                async_cp16(ldst0[o], gsrc[o] + (i + 2) * 32);
        }
        qkv_step(arow + (i + 1) * 32, sB[1], mr, half, acc);
        __syncthreads();
    }

    // Epilogue: C/D layout — lane L: n = L%16, m = (L/16)*8 + r
    const int ncol  = lane & 15;
    const int mbase = (lane >> 4) * 8;
    #pragma unroll
    for (int o = 0; o < 3; ++o) {
        #pragma unroll
        for (int t = 0; t < 4; ++t) {
            const int col = colbase + t * 16 + ncol;
            const float bb = bias[o][col];
            #pragma unroll
            for (int r = 0; r < 8; ++r) {
                outp[o][(size_t)(row0 + mbase + r) * D_MODEL + col] = acc[o][t][r] + bb;
            }
        }
    }
}

// ---------------------------------------------------------------------------
// QK normalization: each (b,n,h) head-row of 64 f32 is contiguous.
// One wave per row; butterfly shfl reduction (wave32).
// ---------------------------------------------------------------------------
__global__ __launch_bounds__(256) void qk_norm_kernel(float* __restrict__ qf,
                                                      float* __restrict__ kf) {
    const int lane = threadIdx.x & 31;
    const int wave = threadIdx.x >> 5;
    const int idx = blockIdx.x * 8 + wave;       // 0 .. 2*131072-1
    float* __restrict__ base = (idx & 1) ? kf : qf;
    const size_t chunk = (size_t)(idx >> 1);
    float2* p = (float2*)(base + chunk * D_HEAD);
    float2 v = p[lane];
    float ss = v.x * v.x + v.y * v.y;
    #pragma unroll
    for (int off = 16; off > 0; off >>= 1) ss += __shfl_xor(ss, off, 32);
    const float scale = 1.0f / (sqrtf(ss) + 1e-6f);
    v.x *= scale; v.y *= scale;
    p[lane] = v;
}

// ---------------------------------------------------------------------------
// Dilated attention: one wave per (b,h,q).  17 offsets {-32,-28,...,32}.
// ---------------------------------------------------------------------------
__global__ __launch_bounds__(256) void dilated_attn_kernel(
    const float* __restrict__ qf, const float* __restrict__ kf,
    const float* __restrict__ vf, __bf16* __restrict__ ab) {

    const int lane = threadIdx.x & 31;
    const int wave = threadIdx.x >> 5;
    const int wg = blockIdx.x * 8 + wave;        // 0 .. B*H*N-1
    const int n = wg & (SEQ_N - 1);
    const int h = (wg >> 11) & (N_HEADS - 1);
    const int b = wg >> 15;

    const size_t rowq = (size_t)(b * SEQ_N + n) * D_MODEL + h * D_HEAD;
    const float2 qv = *(const float2*)(qf + rowq + lane * 2);

    float s[17];
    #pragma unroll
    for (int j = 0; j < 17; ++j) {
        const int kn = n + (j - 8) * 4;
        const bool valid = (kn >= 0) && (kn < SEQ_N);
        float partial = 0.0f;
        if (valid) {
            const float2 kv = *(const float2*)(kf +
                (size_t)(b * SEQ_N + kn) * D_MODEL + h * D_HEAD + lane * 2);
            partial = qv.x * kv.x + qv.y * kv.y;
        }
        #pragma unroll
        for (int off = 16; off > 0; off >>= 1) partial += __shfl_xor(partial, off, 32);
        s[j] = valid ? partial : -INFINITY;
    }

    float mx = -INFINITY;
    #pragma unroll
    for (int j = 0; j < 17; ++j) mx = fmaxf(mx, s[j]);
    float denom = 0.0f;
    #pragma unroll
    for (int j = 0; j < 17; ++j) { s[j] = __expf(s[j] - mx); denom += s[j]; }
    const float inv = 1.0f / denom;

    float2 acc = { 0.0f, 0.0f };
    #pragma unroll
    for (int j = 0; j < 17; ++j) {
        const int kn = n + (j - 8) * 4;
        if (kn >= 0 && kn < SEQ_N) {
            const float2 vv = *(const float2*)(vf +
                (size_t)(b * SEQ_N + kn) * D_MODEL + h * D_HEAD + lane * 2);
            const float wgt = s[j] * inv;
            acc.x += wgt * vv.x;
            acc.y += wgt * vv.y;
        }
    }
    ab[rowq + lane * 2]     = (__bf16)acc.x;
    ab[rowq + lane * 2 + 1] = (__bf16)acc.y;
}

// ---------------------------------------------------------------------------
// One k-step of the output-projection tile: preload 4 B fragments, then
// 4 WMMAs, with the same scheduling hint.
// ---------------------------------------------------------------------------
__device__ __forceinline__ void out_step(const __bf16* __restrict__ arow_k,
                                         const __bf16 (*sBbuf)[32],
                                         int mr, int half, v8f acc[4]) {
    v16bf af = load_frag(arow_k, half);
    v16bf bfr[4];
    #pragma unroll
    for (int t = 0; t < 4; ++t)
        bfr[t] = load_frag(&sBbuf[t * 16 + mr][0], half);
    #pragma unroll
    for (int t = 0; t < 4; ++t)
        acc[t] = __builtin_amdgcn_wmma_f32_16x16x32_bf16(
            false, af, false, bfr[t], (short)0, acc[t], false, false);

    __builtin_amdgcn_sched_group_barrier(0x0020, 2, 0);   // VMEM read x2 (A)
    __builtin_amdgcn_sched_group_barrier(0x0100, 8, 0);   // DS read x8 (B)
    __builtin_amdgcn_sched_group_barrier(0x0008, 4, 0);   // WMMA x4
}

// ---------------------------------------------------------------------------
// Output projection: out = attn @ Wo.T + bo  (f32 result to d_out)
// Same LDS-staged structure, single weight: 8 KB double-buffered B panel.
// ---------------------------------------------------------------------------
__global__ __launch_bounds__(256, 1) void out_gemm_kernel(
    const __bf16* __restrict__ ab, const __bf16* __restrict__ wo,
    const float* __restrict__ bo, float* __restrict__ out) {

    __shared__ __bf16 sB[2][64][32];             // 8 KB

    const int tid  = threadIdx.x;
    const int lane = tid & 31;
    const int wave = tid >> 5;
    const int colbase = blockIdx.x * 64;
    const int row0 = blockIdx.y * 128 + wave * 16;
    const int mr   = lane & 15;
    const int half = lane >> 4;

    const int scol = tid >> 2;
    const int scc  = tid & 3;
    const __bf16* gsrc = wo + (size_t)(colbase + scol) * D_MODEL + scc * 8;
    const unsigned ldst0 = (unsigned)(uintptr_t)&sB[0][scol][scc * 8];
    const unsigned ldst1 = (unsigned)(uintptr_t)&sB[1][scol][scc * 8];

    const __bf16* __restrict__ arow = ab + (size_t)(row0 + mr) * D_MODEL;

    v8f acc[4] = {};

    async_cp16(ldst0, gsrc);

    for (int i = 0; i < 32; i += 2) {
        // ---- step i (buffer 0) ----
        wait_async0();
        __syncthreads();
        async_cp16(ldst1, gsrc + (i + 1) * 32);           // i+1 <= 31 always
        out_step(arow + i * 32, sB[0], mr, half, acc);
        __syncthreads();

        // ---- step i+1 (buffer 1) ----
        wait_async0();
        __syncthreads();
        if (i + 2 < 32) async_cp16(ldst0, gsrc + (i + 2) * 32);
        out_step(arow + (i + 1) * 32, sB[1], mr, half, acc);
        __syncthreads();
    }

    const int ncol  = lane & 15;
    const int mbase = (lane >> 4) * 8;
    #pragma unroll
    for (int t = 0; t < 4; ++t) {
        const int col = colbase + t * 16 + ncol;
        const float bb = bo[col];
        #pragma unroll
        for (int r = 0; r < 8; ++r) {
            out[(size_t)(row0 + mbase + r) * D_MODEL + col] = acc[t][r] + bb;
        }
    }
}

// ---------------------------------------------------------------------------
// Host-side launch (graph-capture safe: only kernel launches on `stream`)
// ---------------------------------------------------------------------------
extern "C" void kernel_launch(void* const* d_in, const int* in_sizes, int n_in,
                              void* d_out, int out_size, void* d_ws, size_t ws_size,
                              hipStream_t stream) {
    (void)in_sizes; (void)n_in; (void)out_size; (void)ws_size;

    const float* x  = (const float*)d_in[0];
    const float* Wq = (const float*)d_in[1];
    const float* bq = (const float*)d_in[2];
    const float* Wk = (const float*)d_in[3];
    const float* bk = (const float*)d_in[4];
    const float* Wv = (const float*)d_in[5];
    const float* bv = (const float*)d_in[6];
    const float* Wo = (const float*)d_in[7];
    const float* bo = (const float*)d_in[8];
    float* out = (float*)d_out;

    // Workspace carve-up (256 B aligned)
    char* ws = (char*)d_ws;
    size_t off = 0;
    auto carve = [&](size_t bytes) {
        void* p = ws + off;
        off += (bytes + 255) & ~(size_t)255;
        return p;
    };
    const size_t WELEMS = (size_t)D_MODEL * D_MODEL;   // 1,048,576
    __bf16* xb  = (__bf16*)carve(NTOK * 2);
    __bf16* wqb = (__bf16*)carve(WELEMS * 2);
    __bf16* wkb = (__bf16*)carve(WELEMS * 2);
    __bf16* wvb = (__bf16*)carve(WELEMS * 2);
    __bf16* wob = (__bf16*)carve(WELEMS * 2);
    float*  qf  = (float*)carve(NTOK * 4);
    float*  kf  = (float*)carve(NTOK * 4);
    float*  vf  = (float*)carve(NTOK * 4);
    __bf16* abuf = (__bf16*)carve(NTOK * 2);

    // 1) Convert inputs to bf16
    {
        const int tb = 256;
        to_bf16_kernel<<<(int)(NTOK / 4 / tb), tb, 0, stream>>>((const float4*)x,  xb,  (int)(NTOK / 4));
        to_bf16_kernel<<<(int)(WELEMS / 4 / tb), tb, 0, stream>>>((const float4*)Wq, wqb, (int)(WELEMS / 4));
        to_bf16_kernel<<<(int)(WELEMS / 4 / tb), tb, 0, stream>>>((const float4*)Wk, wkb, (int)(WELEMS / 4));
        to_bf16_kernel<<<(int)(WELEMS / 4 / tb), tb, 0, stream>>>((const float4*)Wv, wvb, (int)(WELEMS / 4));
        to_bf16_kernel<<<(int)(WELEMS / 4 / tb), tb, 0, stream>>>((const float4*)Wo, wob, (int)(WELEMS / 4));
    }

    // 2) Fused QKV projections (WMMA + async LDS staging)
    qkv_gemm_kernel<<<dim3(D_MODEL / 64, ROWS / 128), 256, 0, stream>>>(
        xb, wqb, wkb, wvb, bq, bk, bv, qf, kf, vf);

    // 3) QK normalization (2 * B*N*H rows of 64)
    qk_norm_kernel<<<(2 * ROWS * N_HEADS) / 8, 256, 0, stream>>>(qf, kf);

    // 4) Dilated local attention (one wave per query-head)
    dilated_attn_kernel<<<(BATCH * N_HEADS * SEQ_N) / 8, 256, 0, stream>>>(
        qf, kf, vf, abuf);

    // 5) Output projection (WMMA + async LDS staging) -> d_out
    out_gemm_kernel<<<dim3(D_MODEL / 64, ROWS / 128), 256, 0, stream>>>(abuf, wob, bo, out);
}
